// Model_17059610099809
// MI455X (gfx1250) — matrix operand order repaired and checked by reading, not compiled, not run on hardware
//
#include <hip/hip_runtime.h>

typedef __attribute__((ext_vector_type(16))) _Float16 v16h;
typedef __attribute__((ext_vector_type(8)))  _Float16 v8h;
typedef __attribute__((ext_vector_type(8)))  float    v8f;

#define BATCH 4096

// ---------------------------------------------------------------- fast transcendentals (gfx1250 TANH trans op)
#if __has_builtin(__builtin_amdgcn_tanh_f32)
  #define FAST_TANH __builtin_amdgcn_tanh_f32
#elif __has_builtin(__builtin_amdgcn_tanhf)
  #define FAST_TANH __builtin_amdgcn_tanhf
#else
  #define FAST_TANH tanhf
#endif
__device__ __forceinline__ float sigm(float x){ return 0.5f * FAST_TANH(0.5f * x) + 0.5f; }

// ---------------------------------------------------------------- async global->LDS staging (ASYNCcnt path)
__device__ __forceinline__ void async_stage_b128(const void* gsrc, void* lds_dst){
  unsigned int  ldsoff = (unsigned int)(uintptr_t)lds_dst;      // low 32 bits = LDS offset
  unsigned long long ga = (unsigned long long)(uintptr_t)gsrc;
  asm volatile("global_load_async_to_lds_b128 %0, %1, off"
               :: "v"(ldsoff), "v"(ga)
               : "memory");
}
__device__ __forceinline__ void wait_async0(){
#if __has_builtin(__builtin_amdgcn_s_wait_asynccnt)
  __builtin_amdgcn_s_wait_asynccnt(0);
#else
  asm volatile("s_wait_asynccnt 0" ::: "memory");
#endif
}

// ---------------------------------------------------------------- WMMA helpers
__device__ __forceinline__ v16h mk16(v8h lo, v8h hi){
  v16h r;
#pragma unroll
  for (int i = 0; i < 8; ++i){ r[i] = lo[i]; r[i+8] = hi[i]; }
  return r;
}

// A fragment: 16xK row-major tile (row stride ld halves), caller applies k-tile offset.
// ISA layout: lanes 0-15 hold M=lane, K 0..7 & 16..23 ; lanes 16-31 hold K 8..15 & 24..31.
__device__ __forceinline__ v16h load_a16(const _Float16* p0, int ld){
  int lane = threadIdx.x & 31;
  const _Float16* p = p0 + (lane & 15) * ld + ((lane & 16) >> 1);
  return mk16(*(const v8h*)p, *(const v8h*)(p + 16));
}

// B fragment from weight stored row-major [N][K]; w0 points at (n0,k0); B[k][n] = W[n][k].
__device__ __forceinline__ v16h load_b16(const _Float16* w0, int ld){
  int lane = threadIdx.x & 31;
  const _Float16* p = w0 + (lane & 15) * ld + ((lane & 16) >> 1);
  return mk16(*(const v8h*)p, *(const v8h*)(p + 16));
}

__device__ __forceinline__ v8f wmma_f16(v16h a, v16h b, v8f c){
  return __builtin_amdgcn_wmma_f32_16x16x32_f16(false, a, false, b, (short)0, c, false, false);
}

// ---------------------------------------------------------------- group tables
// groups 0..29: stream st (0..5) step s -> g = st*5+s ; group 30: q_sum
__constant__ int g_base[31] = {
  211,224,237,250,263,      // s1: user_item_seq[-65:], chunk 13, 10 rows
  662,672,682,692,702,      // s2: user_item_query[:50], chunk 10, 5 rows
  612,622,632,642,652,      // s3: user_query_item[-50:], chunk 10, 5 rows
  397,410,423,436,449,      // s4: user_query_seq[-65:], chunk 13, 10 rows
  512,522,532,542,552,      // c1: query_item_query[-50:], chunk 10, 5 rows
  862,872,882,892,902,      // c2: query_user_item[-50:], chunk 10, 5 rows
  276                       // q_sum: query_feat[:10]
};
__constant__ int g_rows[31] = {
  10,10,10,10,10, 5,5,5,5,5, 5,5,5,5,5, 10,10,10,10,10, 5,5,5,5,5, 5,5,5,5,5, 10
};

// ---------------------------------------------------------------- prep kernels
__global__ void cvt_f16_kernel(const float* __restrict__ src, _Float16* __restrict__ dst, int n){
  int i = blockIdx.x * 256 + threadIdx.x;
  if (i < n) dst[i] = (_Float16)src[i];
}

// Permute LSTM gate rows so tile nt = 4*ug + gate, lane-slot = unit%16, ug = unit/16.
__global__ void lstm_prep_kernel(const float* __restrict__ wih, const float* __restrict__ whh,
                                 const float* __restrict__ bih, const float* __restrict__ bhh,
                                 _Float16* __restrict__ dwih, _Float16* __restrict__ dwhh,
                                 float* __restrict__ dbias){
  int blk = blockIdx.x;            // 0..1023
  int st = blk >> 8, r = blk & 255;
  int g = r >> 6, j = r & 63;
  int nr = (j & 15) + 16 * g + 64 * (j >> 4);
  const float* s1 = wih + (size_t)(st*256 + r) * 64;
  const float* s2 = whh + (size_t)(st*256 + r) * 64;
  _Float16* d1 = dwih + (size_t)(st*256 + nr) * 64;
  _Float16* d2 = dwhh + (size_t)(st*256 + nr) * 64;
  for (int e = threadIdx.x; e < 64; e += 64){
    d1[e] = (_Float16)s1[e];
    d2[e] = (_Float16)s2[e];
  }
  if (threadIdx.x == 0) dbias[st*256 + nr] = bih[st*256 + r] + bhh[st*256 + r];
}

// ---------------------------------------------------------------- mask length (n = sum(sign))
__global__ __launch_bounds__(256) void count_kernel(const float* __restrict__ x, float* __restrict__ Lv){
  __shared__ float red[256];
  int g = blockIdx.x;
  int base = g_base[g], rows = g_rows[g];
  float s = 0.f;
  int total = rows * BATCH;
  const float* p = x + (size_t)base * BATCH;   // rows are contiguous
  for (int i = threadIdx.x; i < total; i += 256){
    float v = p[i];
    s += (v > 0.f) ? 1.f : ((v < 0.f) ? -1.f : 0.f);
  }
  red[threadIdx.x] = s;
  __syncthreads();
  for (int o = 128; o > 0; o >>= 1){
    if (threadIdx.x < o) red[threadIdx.x] += red[threadIdx.x + o];
    __syncthreads();
  }
  if (threadIdx.x == 0){
    float n = red[0];
    float L = (n > 1.f) ? n : 1.f;
    float fr = (float)rows;
    Lv[g] = (L < fr) ? L : fr;
  }
}

// ---------------------------------------------------------------- embedding masked means (HBM-bound gathers)
__global__ __launch_bounds__(256) void means_kernel(const float* __restrict__ x,
                                                    const float* __restrict__ emb,
                                                    const float* __restrict__ Lv,
                                                    _Float16* __restrict__ seq,
                                                    _Float16* __restrict__ feat){
  int col = blockIdx.x * 256 + threadIdx.x;
  int g = blockIdx.y;
  float L = Lv[g];
  int base = g_base[g], rows = g_rows[g];
  float acc[64];
#pragma unroll
  for (int e = 0; e < 64; ++e) acc[e] = 0.f;
  int idx = (int)x[(size_t)base * BATCH + col];
  for (int r = 0; r < rows; ++r){
    if ((float)r >= L) break;            // prefix mask
    int nidx = idx;
    if (r + 1 < rows){
      nidx = (int)x[(size_t)(base + r + 1) * BATCH + col];
      __builtin_prefetch(emb + (size_t)nidx * 64, 0, 0);   // hide gather latency
    }
    const float4* ep = (const float4*)(emb + (size_t)idx * 64);
#pragma unroll
    for (int q = 0; q < 16; ++q){
      float4 v = ep[q];
      acc[4*q+0] += v.x; acc[4*q+1] += v.y; acc[4*q+2] += v.z; acc[4*q+3] += v.w;
    }
    idx = nidx;
  }
  float inv = 1.f / L;
  _Float16* dst = (g < 30) ? (seq + ((size_t)g * BATCH + col) * 64)
                           : (feat + (size_t)col * 448);   // q_sum -> feat cols 0..63
#pragma unroll
  for (int e = 0; e < 64; ++e) dst[e] = (_Float16)(acc[e] * inv);
}

// ---------------------------------------------------------------- wide path (VALU, tiny)
__global__ __launch_bounds__(256) void wide_kernel(const float* __restrict__ x,
                                                   const float* __restrict__ ww,
                                                   const float* __restrict__ wb,
                                                   _Float16* __restrict__ qwf){
  __shared__ float sW[64*81];
  __shared__ float sB[64];
  for (int i = threadIdx.x; i < 64*81; i += 256) sW[i] = ww[i];
  if (threadIdx.x < 64) sB[threadIdx.x] = wb[threadIdx.x];
  __syncthreads();
  int col = blockIdx.x * 256 + threadIdx.x;
  float acc[64];
#pragma unroll
  for (int o = 0; o < 64; ++o) acc[o] = 0.f;
  for (int r = 0; r < 81; ++r){
    float xv = x[(size_t)r * BATCH + col];
#pragma unroll
    for (int o = 0; o < 64; ++o) acc[o] += sW[o*81 + r] * xv;
  }
#pragma unroll
  for (int o = 0; o < 64; ++o)
    qwf[(size_t)col * 128 + 64 + o] = (_Float16)FAST_TANH(acc[o] + sB[o]);
}

// ---------------------------------------------------------------- LSTM (WMMA, per-16-row wave tile)
__global__ __launch_bounds__(128) void lstm_kernel(const _Float16* __restrict__ seq,
                                                   const _Float16* __restrict__ wih,
                                                   const _Float16* __restrict__ whh,
                                                   const float* __restrict__ biasc,
                                                   const float* __restrict__ h0,
                                                   const float* __restrict__ c0,
                                                   const _Float16* __restrict__ wl,
                                                   const float* __restrict__ bl,
                                                   _Float16* __restrict__ feat){
  __shared__ __align__(16) _Float16 sWhh[256*64];   // 32 KB: recurrent weights (hot every step)
  __shared__ __align__(16) _Float16 sH[4][16*64];   // per-wave h buffer
  const int st   = blockIdx.y;
  const int tid  = threadIdx.x;
  const int wave = tid >> 5, lane = tid & 31;
  const int rb   = blockIdx.x * 64 + wave * 16;     // batch row base of this wave
  const int rO   = (lane & 16) >> 1;                // +8 rows for upper half-wave (D layout)

  // async stage of Whh (no VGPR round trip, ASYNCcnt-tracked)
  {
    const uint4* g1 = (const uint4*)(whh + (size_t)st * 256 * 64);
    uint4* d1 = (uint4*)sWhh;
    for (int i = tid; i < 2048; i += 128) async_stage_b128(g1 + i, d1 + i);
  }
  __builtin_prefetch(wih + (size_t)st * 256 * 64, 0, 3);   // warm L2/WGP$ for streamed Wih frags
  // init h (f16, LDS) and c (registers)
  _Float16* myH = sH[wave];
  const float* h0p = h0 + (size_t)st * BATCH * 64 + (size_t)rb * 64;
  for (int i = lane; i < 16*64; i += 32) myH[i] = (_Float16)h0p[i];
  v8f creg[4];
  {
    const float* c0p = c0 + (size_t)st * BATCH * 64;
#pragma unroll
    for (int ug = 0; ug < 4; ++ug){
      int unit = ug*16 + (lane & 15);
#pragma unroll
      for (int j = 0; j < 8; ++j)
        creg[ug][j] = c0p[(size_t)(rb + j + rO) * 64 + unit];
    }
  }
  wait_async0();
  __syncthreads();

  const _Float16* gWih = wih + (size_t)st * 256 * 64;
  const float*    bp   = biasc + st * 256;

#pragma unroll 1   // serial recurrence: keep one step's fragments live (VGPR pressure)
  for (int t = 0; t < 5; ++t){
    const _Float16* xb = seq + ((size_t)(st*5 + t) * BATCH + rb) * 64;
    v16h ax0 = load_a16(xb,        64);
    v16h ax1 = load_a16(xb + 32,   64);
    v16h ah0 = load_a16(myH,       64);
    v16h ah1 = load_a16(myH + 32,  64);
#pragma unroll
    for (int ug = 0; ug < 4; ++ug){
      v8f acc[4];
#pragma unroll
      for (int g = 0; g < 4; ++g){
        int nt = ug*4 + g;
        float bv = bp[nt*16 + (lane & 15)];
#pragma unroll
        for (int j = 0; j < 8; ++j) acc[g][j] = bv;
        acc[g] = wmma_f16(ax0, load_b16(gWih + nt*16*64,      64), acc[g]);
        acc[g] = wmma_f16(ax1, load_b16(gWih + nt*16*64 + 32, 64), acc[g]);
        acc[g] = wmma_f16(ah0, load_b16(sWhh + nt*16*64,      64), acc[g]);
        acc[g] = wmma_f16(ah1, load_b16(sWhh + nt*16*64 + 32, 64), acc[g]);
      }
      int unit = ug*16 + (lane & 15);
#pragma unroll
      for (int j = 0; j < 8; ++j){
        float iv = sigm(acc[0][j]);
        float fv = sigm(acc[1][j]);
        float gv = FAST_TANH(acc[2][j]);
        float ov = sigm(acc[3][j]);
        float cn = fv * creg[ug][j] + iv * gv;
        creg[ug][j] = cn;
        myH[(j + rO)*64 + unit] = (_Float16)(ov * FAST_TANH(cn));   // same-wave DS is in-order
      }
    }
  }

  // s = tanh(h @ w_l^T + b_l)  ->  feat columns 64 + 64*st
  {
    const _Float16* gWl = wl + (size_t)st * 64 * 64;
    const float*    blp = bl + st * 64;
    v16h ah0 = load_a16(myH,      64);
    v16h ah1 = load_a16(myH + 32, 64);
#pragma unroll
    for (int nt = 0; nt < 4; ++nt){
      float bv = blp[nt*16 + (lane & 15)];
      v8f acc;
#pragma unroll
      for (int j = 0; j < 8; ++j) acc[j] = bv;
      acc = wmma_f16(ah0, load_b16(gWl + nt*16*64,      64), acc);
      acc = wmma_f16(ah1, load_b16(gWl + nt*16*64 + 32, 64), acc);
      int coloff = 64 + st*64 + nt*16 + (lane & 15);
#pragma unroll
      for (int j = 0; j < 8; ++j)
        feat[(size_t)(rb + j + rO) * 448 + coloff] = (_Float16)FAST_TANH(acc[j]);
    }
  }
}

// ---------------------------------------------------------------- CNN streams (tiny, VALU)
__global__ __launch_bounds__(256) void cnn_kernel(const _Float16* __restrict__ seq,
                                                  const _Float16* __restrict__ c1w, const float* __restrict__ c1b,
                                                  const _Float16* __restrict__ l1w, const float* __restrict__ l1b,
                                                  const _Float16* __restrict__ c2w, const float* __restrict__ c2b,
                                                  const _Float16* __restrict__ l2w, const float* __restrict__ l2b,
                                                  _Float16* __restrict__ feat){
  int cs = blockIdx.y;   // 0 = c1, 1 = c2
  __shared__ __align__(16) _Float16 sCw[12*128];
  __shared__ __align__(16) _Float16 sLw[64*24];
  __shared__ float sCb[12], sLb[64];
  {
    const _Float16* cw = cs ? c2w : c1w;
    const _Float16* lw = cs ? l2w : l1w;
    const float*    cb = cs ? c2b : c1b;
    const float*    lb = cs ? l2b : l1b;
    for (int i = threadIdx.x; i < 12*128; i += 256) sCw[i] = cw[i];
    for (int i = threadIdx.x; i < 64*24;  i += 256) sLw[i] = lw[i];
    if (threadIdx.x < 12) sCb[threadIdx.x] = cb[threadIdx.x];
    if (threadIdx.x < 64) sLb[threadIdx.x] = lb[threadIdx.x];
  }
  __syncthreads();
  int col = blockIdx.x * 256 + threadIdx.x;
  const size_t sbase = (size_t)((4 + cs) * 5) * BATCH * 64;

  float y[12][4];
#pragma unroll
  for (int oc = 0; oc < 12; ++oc){
#pragma unroll
    for (int t = 0; t < 4; ++t){
      float a = sCb[oc];
#pragma unroll
      for (int dt = 0; dt < 2; ++dt){
        const _Float16* m = seq + sbase + ((size_t)(t + dt) * BATCH + col) * 64;
        const _Float16* w = sCw + oc*128 + dt*64;
#pragma unroll
        for (int q = 0; q < 8; ++q){
          v8h mv = *(const v8h*)(m + 8*q);
          v8h wv = *(const v8h*)(w + 8*q);
#pragma unroll
          for (int i = 0; i < 8; ++i) a += (float)mv[i] * (float)wv[i];
        }
      }
      y[oc][t] = fmaxf(a, 0.f);
    }
  }
  float p[24];
#pragma unroll
  for (int oc = 0; oc < 12; ++oc){
    p[0*12 + oc] = fmaxf(y[oc][0], y[oc][1]);
    p[1*12 + oc] = fmaxf(y[oc][2], y[oc][3]);
  }
  int coloff = 320 + cs*64;
#pragma unroll
  for (int o = 0; o < 64; ++o){
    float a = sLb[o];
#pragma unroll
    for (int k = 0; k < 24; ++k) a += p[k] * (float)sLw[o*24 + k];
    feat[(size_t)col * 448 + coloff + o] = (_Float16)fmaxf(a, 0.f);
  }
}

// ---------------------------------------------------------------- q = tanh(feat @ cq_w^T + b) (WMMA, K=448)
__global__ __launch_bounds__(128) void cq_kernel(const _Float16* __restrict__ feat,
                                                 const _Float16* __restrict__ cqw,
                                                 const float* __restrict__ cqb,
                                                 _Float16* __restrict__ qwf){
  __shared__ __align__(16) _Float16 sW[64*448];     // 56 KB
  __shared__ float sB[64];
  for (int i = threadIdx.x; i < 64*448/8; i += 128)
    async_stage_b128(((const uint4*)cqw) + i, ((uint4*)sW) + i);
  if (threadIdx.x < 64) sB[threadIdx.x] = cqb[threadIdx.x];
  wait_async0();
  __syncthreads();
  const int wave = threadIdx.x >> 5, lane = threadIdx.x & 31;
  const int rb = blockIdx.x * 64 + wave * 16;
  const int rO = (lane & 16) >> 1;
  const _Float16* arow = feat + (size_t)rb * 448;
#pragma unroll
  for (int nt = 0; nt < 4; ++nt){
    float bv = sB[nt*16 + (lane & 15)];
    v8f acc;
#pragma unroll
    for (int j = 0; j < 8; ++j) acc[j] = bv;
#pragma unroll
    for (int kt = 0; kt < 14; ++kt)
      acc = wmma_f16(load_a16(arow + kt*32, 448), load_b16(sW + nt*16*448 + kt*32, 448), acc);
#pragma unroll
    for (int j = 0; j < 8; ++j)
      qwf[(size_t)(rb + j + rO) * 128 + nt*16 + (lane & 15)] = (_Float16)FAST_TANH(acc[j]);
  }
}

// ---------------------------------------------------------------- qw = tanh(qwf @ cqu_w^T + b); out = dw_w @ qw + dw_b
__global__ __launch_bounds__(128) void cqu_out_kernel(const _Float16* __restrict__ qwf,
                                                      const _Float16* __restrict__ cquw,
                                                      const float* __restrict__ cqub,
                                                      const float* __restrict__ dww,
                                                      const float* __restrict__ dwb,
                                                      float* __restrict__ out){
  __shared__ __align__(16) _Float16 sW[64*128];     // 16 KB
  __shared__ float sB[64], sDw[64];
  __shared__ float sQ[4][16*64];
  for (int i = threadIdx.x; i < 64*128/8; i += 128)
    async_stage_b128(((const uint4*)cquw) + i, ((uint4*)sW) + i);
  if (threadIdx.x < 64){ sB[threadIdx.x] = cqub[threadIdx.x]; sDw[threadIdx.x] = dww[threadIdx.x]; }
  wait_async0();
  __syncthreads();
  const int wave = threadIdx.x >> 5, lane = threadIdx.x & 31;
  const int rb = blockIdx.x * 64 + wave * 16;
  const int rO = (lane & 16) >> 1;
  const _Float16* arow = qwf + (size_t)rb * 128;
#pragma unroll
  for (int nt = 0; nt < 4; ++nt){
    float bv = sB[nt*16 + (lane & 15)];
    v8f acc;
#pragma unroll
    for (int j = 0; j < 8; ++j) acc[j] = bv;
#pragma unroll
    for (int kt = 0; kt < 4; ++kt)
      acc = wmma_f16(load_a16(arow + kt*32, 128), load_b16(sW + nt*16*128 + kt*32, 128), acc);
#pragma unroll
    for (int j = 0; j < 8; ++j)
      sQ[wave][(j + rO)*64 + nt*16 + (lane & 15)] = FAST_TANH(acc[j]);   // same-wave DS in-order
  }
  float db = dwb[0];
  if (lane < 16){
    float s = db;
#pragma unroll
    for (int jj = 0; jj < 64; ++jj) s += sQ[wave][lane*64 + jj] * sDw[jj];
    out[rb + lane] = s;
  }
}

// ---------------------------------------------------------------- launch
extern "C" void kernel_launch(void* const* d_in, const int* in_sizes, int n_in,
                              void* d_out, int out_size, void* d_ws, size_t ws_size,
                              hipStream_t stream) {
  const float* x       = (const float*)d_in[0];
  const float* embed   = (const float*)d_in[1];
  const float* lWih    = (const float*)d_in[2];
  const float* lWhh    = (const float*)d_in[3];
  const float* lbih    = (const float*)d_in[4];
  const float* lbhh    = (const float*)d_in[5];
  const float* h0      = (const float*)d_in[6];
  const float* c0      = (const float*)d_in[7];
  const float* w_l     = (const float*)d_in[8];
  const float* b_l     = (const float*)d_in[9];
  const float* conv1_w = (const float*)d_in[10];
  const float* conv1_b = (const float*)d_in[11];
  const float* lin1_w  = (const float*)d_in[12];
  const float* lin1_b  = (const float*)d_in[13];
  const float* conv2_w = (const float*)d_in[14];
  const float* conv2_b = (const float*)d_in[15];
  const float* lin2_w  = (const float*)d_in[16];
  const float* lin2_b  = (const float*)d_in[17];
  const float* wide_w  = (const float*)d_in[18];
  const float* wide_b  = (const float*)d_in[19];
  const float* cq_w    = (const float*)d_in[20];
  const float* cq_b    = (const float*)d_in[21];
  const float* cqu_w   = (const float*)d_in[22];
  const float* cqu_b   = (const float*)d_in[23];
  const float* dw_w    = (const float*)d_in[24];
  const float* dw_b    = (const float*)d_in[25];
  float* out = (float*)d_out;

  char* w = (char*)d_ws;
  size_t off = 0;
  auto take = [&](size_t bytes) -> void* {
    void* p = w + off;
    off = (off + bytes + 255) & ~(size_t)255;
    return p;
  };
  float*     Lv     = (float*)    take(31 * 4);
  _Float16*  hWih   = (_Float16*) take((size_t)4*256*64*2);
  _Float16*  hWhh   = (_Float16*) take((size_t)4*256*64*2);
  float*     biasc  = (float*)    take((size_t)4*256*4);
  _Float16*  hWl    = (_Float16*) take((size_t)4*64*64*2);
  _Float16*  hCqw   = (_Float16*) take((size_t)64*448*2);
  _Float16*  hCquw  = (_Float16*) take((size_t)64*128*2);
  _Float16*  hC1w   = (_Float16*) take((size_t)12*128*2);
  _Float16*  hC2w   = (_Float16*) take((size_t)12*128*2);
  _Float16*  hL1w   = (_Float16*) take((size_t)64*24*2);
  _Float16*  hL2w   = (_Float16*) take((size_t)64*24*2);
  _Float16*  seq    = (_Float16*) take((size_t)30*BATCH*64*2);
  _Float16*  feat   = (_Float16*) take((size_t)BATCH*448*2);
  _Float16*  qwf    = (_Float16*) take((size_t)BATCH*128*2);
  (void)ws_size; (void)in_sizes; (void)n_in; (void)out_size;

  // weight prep (f32 -> f16)
  cvt_f16_kernel<<<(4*64*64 + 255)/256, 256, 0, stream>>>(w_l,    hWl,   4*64*64);
  cvt_f16_kernel<<<(64*448  + 255)/256, 256, 0, stream>>>(cq_w,   hCqw,  64*448);
  cvt_f16_kernel<<<(64*128  + 255)/256, 256, 0, stream>>>(cqu_w,  hCquw, 64*128);
  cvt_f16_kernel<<<(12*128  + 255)/256, 256, 0, stream>>>(conv1_w,hC1w,  12*128);
  cvt_f16_kernel<<<(12*128  + 255)/256, 256, 0, stream>>>(conv2_w,hC2w,  12*128);
  cvt_f16_kernel<<<(64*24   + 255)/256, 256, 0, stream>>>(lin1_w, hL1w,  64*24);
  cvt_f16_kernel<<<(64*24   + 255)/256, 256, 0, stream>>>(lin2_w, hL2w,  64*24);
  lstm_prep_kernel<<<1024, 64, 0, stream>>>(lWih, lWhh, lbih, lbhh, hWih, hWhh, biasc);

  // mask lengths, then gathered masked means (q_sum lands in feat cols 0..63)
  count_kernel<<<31, 256, 0, stream>>>(x, Lv);
  means_kernel<<<dim3(BATCH/256, 31), 256, 0, stream>>>(x, embed, Lv, seq, feat);

  // wide path -> qwf cols 64..127
  wide_kernel<<<BATCH/256, 256, 0, stream>>>(x, wide_w, wide_b, qwf);

  // LSTM streams -> feat cols 64..319
  lstm_kernel<<<dim3(BATCH/64, 4), 128, 0, stream>>>(seq, hWih, hWhh, biasc, h0, c0, hWl, b_l, feat);

  // CNN streams -> feat cols 320..447
  cnn_kernel<<<dim3(BATCH/256, 2), 256, 0, stream>>>(seq, hC1w, conv1_b, hL1w, lin1_b,
                                                     hC2w, conv2_b, hL2w, lin2_b, feat);

  // q -> qwf cols 0..63
  cq_kernel<<<BATCH/64, 128, 0, stream>>>(feat, hCqw, cq_b, qwf);

  // qw + final dot -> out
  cqu_out_kernel<<<BATCH/64, 128, 0, stream>>>(qwf, hCquw, cqu_b, dw_w, dw_b, out);
}
